// GNN_51539607552377
// MI455X (gfx1250) — matrix operand order repaired and checked by reading, not compiled, not run on hardware
//
#include <hip/hip_runtime.h>

// ---------------- problem constants ----------------
#define HWPIX 9216              // 96*96
#define TN    128               // pixels per workgroup tile
#define NTB   72                // HWPIX / TN
#define NB    4                 // batch
#define NGRID (NB * NTB)        // 288 workgroups
#define MAXSRC 6
#define XLD   40                // transposed-activation row stride in shorts (80B, 16B-aligned)

typedef __attribute__((ext_vector_type(16))) __bf16         bf16x16;
typedef __attribute__((ext_vector_type(8)))  float          f32x8;
typedef __attribute__((ext_vector_type(4)))  unsigned int   u32x4;
typedef __attribute__((ext_vector_type(8)))  unsigned int   u32x8;

// global (address-space 1) float pointer -> forces global_load, not flat_load
typedef const float __attribute__((address_space(1)))* gptr_t;

__device__ __forceinline__ unsigned short f2bf(float f) {
  union { float f; unsigned int u; } c; c.f = f;
  unsigned int u = c.u + 0x7FFFu + ((c.u >> 16) & 1u);   // RNE
  return (unsigned short)(u >> 16);
}
__device__ __forceinline__ unsigned int pack2bf(float lo, float hi) {
  return (unsigned int)f2bf(lo) | ((unsigned int)f2bf(hi) << 16);
}
__device__ __forceinline__ gptr_t to_gptr(const float* p) {
  return (gptr_t)(unsigned long long)(const void*)p;
}
// Row pointers in the fill loop are wave-uniform by construction; force them into
// SGPRs so global loads use the SADDR form (base in SGPR pair + 32-bit voffset).
__device__ __forceinline__ gptr_t uniform_row(gptr_t p) {
  unsigned long long v = (unsigned long long)p;
  unsigned lo = __builtin_amdgcn_readfirstlane((unsigned)v);
  unsigned hi = __builtin_amdgcn_readfirstlane((unsigned)(v >> 32));
  return (gptr_t)(((unsigned long long)hi << 32) | lo);
}

// Load one lane's 32-byte WMMA fragment as two 16B LDS vector loads (-> ds_load_b128 x2).
__device__ __forceinline__ bf16x16 frag_load2(const unsigned short* p0, const unsigned short* p1) {
  u32x4 q0 = *(const u32x4*)p0;
  u32x4 q1 = *(const u32x4*)p1;
  u32x8 r;
#pragma unroll
  for (int i = 0; i < 4; ++i) { r[i] = q0[i]; r[i + 4] = q1[i]; }
  return __builtin_bit_cast(bf16x16, r);
}

// One 16x16x32 bf16 WMMA: A row-major [M][lda] (lda mult of 32, bf16);
// B transposed [N][XLD] (per-column contiguous K). Layouts per CDNA5 ISA 7.12.2.
__device__ __forceinline__ f32x8 wmma_tile(const unsigned short* sA, int lda, int arow0, int ak0,
                                           const unsigned short* sBT, int bcol0,
                                           f32x8 c) {
  const int lane = threadIdx.x & 31;
  const int half = lane >> 4;
  const int sub  = lane & 15;
  const unsigned short* ap = sA + (arow0 + sub) * lda + ak0 + half * 8;
  bf16x16 a = frag_load2(ap, ap + 16);
  const unsigned short* bp = sBT + (bcol0 + sub) * XLD + half * 8;
  bf16x16 b = frag_load2(bp, bp + 16);
  return __builtin_amdgcn_wmma_f32_16x16x32_bf16(false, a, false, b, (short)0, c, false, false);
}

// ---------------- fused conv block: relu(bn(W2*relu(bn(W1*cat(srcs))))) [+adds] ----------------
struct BlockArgs {
  const float* src[MAXSRC];
  int          srcC[MAXSRC];
  int          nsrc;
  int          cin;                          // sum(srcC), <= 306
  const float *W1, *s1, *b1;                 // [32,cin],[32],[32]
  const float *W2, *s2, *b2;                 // [10,32],[10],[10]
  const float *add0; int add0C, add0off;     // optional fused message adds
  const float *add1; int add1C, add1off;
  const float* zrow;                         // 128 zero floats (pad channels)
  float*       dst;                          // [B,10,HW]
};

__global__ __launch_bounds__(256) void gnn_block_kernel(BlockArgs A) {
  __shared__ alignas(16) unsigned short sW1[32 * 320];   // bf16 W1, K padded to 320
  __shared__ alignas(16) unsigned short sW2[16 * 32];
  __shared__ alignas(16) unsigned short sXT[2][TN * XLD]; // double-buffered activation chunk
  __shared__ alignas(16) unsigned short sYT[TN * XLD];    // mid activations, transposed
  __shared__ gptr_t sRow[320];                            // per-channel global row pointers
  __shared__ float sS1[32], sB1[32], sS2[16], sB2[16];

  const int tid = threadIdx.x, lane = tid & 31, wv = tid >> 5;
  const int b   = blockIdx.x / NTB;
  const int hw0 = (blockIdx.x % NTB) * TN;
  const int Kpad = (A.cin + 31) & ~31;
  const int nk   = Kpad >> 5;

  // bf16 weights into LDS (zero-padded)
  for (int i = tid; i < 32 * Kpad; i += 256) {
    int m = i / Kpad, k = i - m * Kpad;
    sW1[i] = (k < A.cin) ? f2bf(A.W1[m * A.cin + k]) : (unsigned short)0;
  }
  for (int i = tid; i < 16 * 32; i += 256) {
    int m = i >> 5, k = i & 31;
    sW2[i] = (m < 10) ? f2bf(A.W2[m * 32 + k]) : (unsigned short)0;
  }
  // concat resolved ONCE per channel -> global row pointers; pad channels -> zero page
  for (int ch = tid; ch < Kpad; ch += 256) {
    const float* r;
    if (ch < A.cin) {
      int s = 0, base = 0;
      while (ch >= base + A.srcC[s]) { base += A.srcC[s]; ++s; }
      r = A.src[s] + (size_t)(b * A.srcC[s] + (ch - base)) * HWPIX + hw0;
    } else {
      r = A.zrow;
    }
    sRow[ch] = to_gptr(r);
  }
  if (tid < 32) { sS1[tid] = A.s1[tid]; sB1[tid] = A.b1[tid]; }
  if (tid < 16) { sS2[tid] = (tid < 10) ? A.s2[tid] : 0.f;
                  sB2[tid] = (tid < 10) ? A.b2[tid] : 0.f; }

  // stream one K=32 chunk into a transposed LDS tile (wave-uniform rows -> SADDR loads)
  auto fillX = [&](int c, int buf) {
#pragma unroll
    for (int u = 0; u < (16 * TN) / 256; ++u) {
      int i  = tid + u * 256;
      int k2 = i >> 7, n = i & (TN - 1);    // k2 is wave-uniform
      int ch = (c << 5) + (k2 << 1);
      gptr_t r0 = uniform_row(sRow[ch]);
      gptr_t r1 = uniform_row(sRow[ch + 1]);
      float v0 = r0[n];
      float v1 = r1[n];
      ((unsigned int*)sXT[buf])[n * (XLD / 2) + k2] = pack2bf(v0, v1);
    }
  };

  __syncthreads();             // tables ready
  fillX(0, 0);
  __syncthreads();

  f32x8 acc0 = {}, acc1 = {};
  for (int c = 0; c < nk; ++c) {
    const int cur = c & 1;
    if (c + 1 < nk) fillX(c + 1, cur ^ 1);   // overlap next chunk's global loads with WMMA
    acc0 = wmma_tile(sW1, Kpad, 0,  c << 5, sXT[cur], wv << 4, acc0);
    acc1 = wmma_tile(sW1, Kpad, 16, c << 5, sXT[cur], wv << 4, acc1);
    __syncthreads();
  }

  // epilogue 1: BN + ReLU, packed b128 stores into transposed mid buffer
  const int half = lane >> 4, sub = lane & 15;
  const int col = (wv << 4) + sub;
  {
    u32x4 p0, p1;
#pragma unroll
    for (int j = 0; j < 4; ++j) {
      int ra = 2 * j + half * 8;          // rows held by this lane are contiguous
      float y0 = fmaxf(acc0[2 * j]     * sS1[ra]     + sB1[ra],     0.f);
      float y1 = fmaxf(acc0[2 * j + 1] * sS1[ra + 1] + sB1[ra + 1], 0.f);
      p0[j] = pack2bf(y0, y1);
      int rb = 16 + ra;
      float z0 = fmaxf(acc1[2 * j]     * sS1[rb]     + sB1[rb],     0.f);
      float z1 = fmaxf(acc1[2 * j + 1] * sS1[rb + 1] + sB1[rb + 1], 0.f);
      p1[j] = pack2bf(z0, z1);
    }
    *(u32x4*)(sYT + col * XLD + half * 8)      = p0;   // K = half*8 .. +7
    *(u32x4*)(sYT + col * XLD + 16 + half * 8) = p1;   // K = 16+half*8 .. +7
  }
  __syncthreads();

  f32x8 acc2 = {};
  acc2 = wmma_tile(sW2, 32, 0, 0, sYT, wv << 4, acc2);

#pragma unroll
  for (int r = 0; r < 8; ++r) {
    int row = r + half * 8;
    if (row < 10) {
      float z = fmaxf(acc2[r] * sS2[row] + sB2[row], 0.f);
      size_t pix = (size_t)hw0 + col;
      if (A.add0) z += A.add0[(size_t)(b * A.add0C + A.add0off + row) * HWPIX + pix];
      if (A.add1) z += A.add1[(size_t)(b * A.add1C + A.add1off + row) * HWPIX + pix];
      A.dst[(size_t)(b * 10 + row) * HWPIX + pix] = z;
    }
  }
}

// ---------------- gated decomposition: relu(bn(W * (fea * sigmoid(att(g))))) ----------------
struct DecompArgs {
  const float* fea;                    // [B,256,HW]
  const float* gsrc;                   // [B,10,HW]
  const float *attW, *attB;            // [10],[1]
  const float *W, *s, *bb;             // [cout,256],[cout],[cout]
  float*       dst;                    // [B,cout,HW]
  int          cout;                   // 20 or 40 (padded to 48 rows)
};

__global__ __launch_bounds__(256) void gnn_decomp_kernel(DecompArgs A) {
  __shared__ alignas(16) unsigned short sW[48 * 256];
  __shared__ alignas(16) unsigned short sXT[2][TN * XLD];
  __shared__ float sGate[TN];
  __shared__ float sS[48], sB[48];

  const int tid = threadIdx.x, lane = tid & 31, wv = tid >> 5;
  const int b   = blockIdx.x / NTB;
  const int hw0 = (blockIdx.x % NTB) * TN;

  for (int i = tid; i < 48 * 256; i += 256) {
    int m = i >> 8, k = i & 255;
    sW[i] = (m < A.cout) ? f2bf(A.W[m * 256 + k]) : (unsigned short)0;
  }
  if (tid < 48) { sS[tid] = (tid < A.cout) ? A.s[tid]  : 0.f;
                  sB[tid] = (tid < A.cout) ? A.bb[tid] : 0.f; }
  if (tid < TN) {
    float acc = A.attB[0];
#pragma unroll
    for (int c = 0; c < 10; ++c)
      acc += A.attW[c] * A.gsrc[(size_t)(b * 10 + c) * HWPIX + hw0 + tid];
    sGate[tid] = 1.f / (1.f + __expf(-acc));
  }

  const gptr_t base = to_gptr(A.fea + (size_t)(b * 256) * HWPIX + hw0);
  auto fillX = [&](int c, int buf) {
#pragma unroll
    for (int u = 0; u < (16 * TN) / 256; ++u) {
      int i  = tid + u * 256;
      int k2 = i >> 7, n = i & (TN - 1);
      int ch = (c << 5) + (k2 << 1);
      float g  = sGate[n];
      float v0 = base[(size_t)ch       * HWPIX + n] * g;
      float v1 = base[(size_t)(ch + 1) * HWPIX + n] * g;
      ((unsigned int*)sXT[buf])[n * (XLD / 2) + k2] = pack2bf(v0, v1);
    }
  };

  __syncthreads();             // gate ready
  fillX(0, 0);
  __syncthreads();

  f32x8 a0 = {}, a1 = {}, a2 = {};
  for (int c = 0; c < 8; ++c) {             // cin = 256 -> 8 chunks of K=32
    const int cur = c & 1;
    if (c + 1 < 8) fillX(c + 1, cur ^ 1);
    a0 = wmma_tile(sW, 256, 0,  c << 5, sXT[cur], wv << 4, a0);
    a1 = wmma_tile(sW, 256, 16, c << 5, sXT[cur], wv << 4, a1);
    a2 = wmma_tile(sW, 256, 32, c << 5, sXT[cur], wv << 4, a2);
    __syncthreads();
  }

  const int half = lane >> 4, sub = lane & 15;
  const int col = (wv << 4) + sub;
#pragma unroll
  for (int r = 0; r < 8; ++r) {
    int row = r + half * 8;
    float v[3] = { a0[r], a1[r], a2[r] };
#pragma unroll
    for (int mt = 0; mt < 3; ++mt) {
      int rr = mt * 16 + row;
      if (rr < A.cout)
        A.dst[(size_t)(b * A.cout + rr) * HWPIX + hw0 + col] =
            fmaxf(v[mt] * sS[rr] + sB[rr], 0.f);
    }
  }
}

// ---------------- host orchestration ----------------
extern "C" void kernel_launch(void* const* d_in, const int* in_sizes, int n_in,
                              void* d_out, int out_size, void* d_ws, size_t ws_size,
                              hipStream_t stream) {
  (void)in_sizes; (void)n_in; (void)out_size; (void)ws_size;

  const float* xp    = (const float*)d_in[0];
  const float* xh    = (const float*)d_in[1];
  const float* xf    = (const float*)d_in[2];
  const float* p_fea = (const float*)d_in[4];
  const float* h_fea = (const float*)d_in[5];
  const float* Pp    = (const float*)d_in[7];    // params flat, insertion order

  const size_t SL = (size_t)NB * 10 * HWPIX;     // one [B,10,H,W] slice
  const float* xps[6];
  for (int j = 0; j < 6; ++j) xps[j] = xp + j * SL;
  const float* xh0 = xh;
  const float* xh1 = xh + SL;

  size_t off = 0;
  auto take = [&](size_t n) { const float* q = Pp + off; off += n; return q; };
  struct BlkW { const float *W1, *s1, *b1, *W2, *s2, *b2; };
  auto takeBlk = [&](int cin) {
    BlkW w; w.W1 = take((size_t)32 * cin); w.s1 = take(32); w.b1 = take(32);
    w.W2 = take(320); w.s2 = take(10); w.b2 = take(10); return w;
  };
  BlkW w_hf = takeBlk(20),  w_fu = takeBlk(20);
  BlkW w_huph = takeBlk(40), w_hulh = takeBlk(266), w_huup = takeBlk(20);
  BlkW w_hlph = takeBlk(20), w_hllh = takeBlk(266), w_hlup = takeBlk(20);
  const float* pn_W1 = take(6 * 32 * 306); const float* pn_s1 = take(192);
  const float* pn_b1 = take(192);          const float* pn_W2 = take(1920);
  const float* pn_s2 = take(60);           const float* pn_b2 = take(60);
  const float* pu_W1 = take(6 * 32 * 20);  const float* pu_s1 = take(192);
  const float* pu_b1 = take(192);          const float* pu_W2 = take(1920);
  const float* pu_s2 = take(60);           const float* pu_b2 = take(60);
  const float* fa_W = take(10); const float* fa_b = take(1);
  const float* ua_W = take(10); const float* ua_b = take(1);
  const float* la_W = take(10); const float* la_b = take(1);
  const float* fd_W = take(5120);  const float* fd_s = take(20); const float* fd_b = take(20);
  const float* ud_W = take(10240); const float* ud_s = take(40); const float* ud_b = take(40);
  const float* ld_W = take(5120);  const float* ld_s = take(20); const float* ld_b = take(20);

  float* O = (float*)d_out;
  float* o_xp_new  = O;
  float* o_xh_new  = O + 6 * SL;
  float* o_xf_new  = O + 8 * SL;
  float* o_xf_msg  = O + 9 * SL;
  float* o_xh_msgs = O + 10 * SL;
  float* o_xp_msgs = O + 12 * SL;

  float* Wk    = (float*)d_ws;
  float* s_xfh = Wk;                   // [B,20,HW]
  float* s_up  = s_xfh + 2 * SL;       // [B,40,HW]
  float* s_lo  = s_up  + 4 * SL;       // [B,20,HW]
  float* s_xlh = s_lo  + 2 * SL;       // [B,10,HW]
  float* s_xuh = s_xlh + SL;           // [B,10,HW]
  float* s_zero = s_xuh + SL;          // 128-float zero page for padded channels
  hipMemsetAsync(s_zero, 0, TN * sizeof(float), stream);   // capture-safe

  dim3 grid(NGRID), blk(256);
  auto LB = [&](BlockArgs& A) {
    A.zrow = s_zero;
    gnn_block_kernel<<<grid, blk, 0, stream>>>(A);
  };
  auto setW = [](BlockArgs& A, const BlkW& w) {
    A.W1 = w.W1; A.s1 = w.s1; A.b1 = w.b1; A.W2 = w.W2; A.s2 = w.s2; A.b2 = w.b2;
  };

  // decompositions (gated convs)
  gnn_decomp_kernel<<<grid, blk, 0, stream>>>(DecompArgs{h_fea, xf,  fa_W, fa_b, fd_W, fd_s, fd_b, s_xfh, 20});
  gnn_decomp_kernel<<<grid, blk, 0, stream>>>(DecompArgs{p_fea, xh0, ua_W, ua_b, ud_W, ud_s, ud_b, s_up,  40});
  gnn_decomp_kernel<<<grid, blk, 0, stream>>>(DecompArgs{p_fea, xh1, la_W, la_b, ld_W, ld_s, ld_b, s_lo,  20});

  // --- full graph ---
  { BlockArgs A{}; A.src[0]=xh0; A.srcC[0]=10; A.src[1]=xh1; A.srcC[1]=10;
    A.nsrc=2; A.cin=20; setW(A, w_hf); A.dst=o_xf_msg; LB(A); }
  { BlockArgs A{}; A.src[0]=xf; A.srcC[0]=10; A.src[1]=o_xf_msg; A.srcC[1]=10;
    A.nsrc=2; A.cin=20; setW(A, w_fu); A.dst=o_xf_new; LB(A); }

  // --- half upper ---
  { BlockArgs A{}; A.src[0]=h_fea; A.srcC[0]=256; A.src[1]=xh1; A.srcC[1]=10;
    A.nsrc=2; A.cin=266; setW(A, w_hulh); A.dst=s_xlh; LB(A); }
  { BlockArgs A{}; for (int i=0;i<4;++i){ A.src[i]=xps[i]; A.srcC[i]=10; }
    A.nsrc=4; A.cin=40; setW(A, w_huph);
    A.add0=s_xlh; A.add0C=10; A.add0off=0;
    A.add1=s_xfh; A.add1C=20; A.add1off=0;           // xfh chunk 0
    A.dst=o_xh_msgs; LB(A); }                        // msg_u
  { BlockArgs A{}; A.src[0]=xh0; A.srcC[0]=10; A.src[1]=o_xh_msgs; A.srcC[1]=10;
    A.nsrc=2; A.cin=20; setW(A, w_huup); A.dst=o_xh_new; LB(A); }

  // --- half lower ---
  { BlockArgs A{}; A.src[0]=h_fea; A.srcC[0]=256; A.src[1]=xh0; A.srcC[1]=10;
    A.nsrc=2; A.cin=266; setW(A, w_hllh); A.dst=s_xuh; LB(A); }
  { BlockArgs A{}; A.src[0]=xps[4]; A.srcC[0]=10; A.src[1]=xps[5]; A.srcC[1]=10;
    A.nsrc=2; A.cin=20; setW(A, w_hlph);
    A.add0=s_xuh; A.add0C=10; A.add0off=0;
    A.add1=s_xfh; A.add1C=20; A.add1off=10;          // xfh chunk 1
    A.dst=o_xh_msgs + SL; LB(A); }                   // msg_l
  { BlockArgs A{}; A.src[0]=xh1; A.srcC[0]=10; A.src[1]=o_xh_msgs + SL; A.srcC[1]=10;
    A.nsrc=2; A.cin=20; setW(A, w_hlup); A.dst=o_xh_new + SL; LB(A); }

  // --- part graph ---
  for (int j = 0; j < 6; ++j) {
    BlkW wn{ pn_W1 + (size_t)j * 32 * 306, pn_s1 + j * 32, pn_b1 + j * 32,
             pn_W2 + j * 320, pn_s2 + j * 10, pn_b2 + j * 10 };
    BlkW wu{ pu_W1 + (size_t)j * 32 * 20,  pu_s1 + j * 32, pu_b1 + j * 32,
             pu_W2 + j * 320, pu_s2 + j * 10, pu_b2 + j * 10 };

    BlockArgs An{};
    An.src[0] = p_fea; An.srcC[0] = 256;
    int s = 1;
    for (int i = 0; i < 6; ++i) if (i != j) { An.src[s] = xps[i]; An.srcC[s] = 10; ++s; }
    An.nsrc = 6; An.cin = 306; setW(An, wn);
    if (j < 4) { An.add0 = s_up; An.add0C = 40; An.add0off = 10 * j; }
    else       { An.add0 = s_lo; An.add0C = 20; An.add0off = 10 * (j - 4); }
    An.dst = o_xp_msgs + j * SL;                     // msg_j
    LB(An);

    BlockArgs Au{};
    Au.src[0] = xps[j];           Au.srcC[0] = 10;
    Au.src[1] = o_xp_msgs + j*SL; Au.srcC[1] = 10;
    Au.nsrc = 2; Au.cin = 20; setW(Au, wu);
    Au.dst = o_xp_new + j * SL;
    LB(Au);
  }
}
